// StarTransformer_11175504904796
// MI455X (gfx1250) — compile-verified, matrix operand order checked by reading, not run on hardware
//
#include <hip/hip_runtime.h>
#include <hip/hip_bf16.h>

// ---------------------------------------------------------------------------
// StarTransformer forward for MI455X (gfx1250, wave32, WMMA).
// Heavy GEMMs: v_wmma_f32_16x16x32_bf16, bf16-preconverted transposed weights,
// 128x128 block tile, double-buffered LDS, v_perm-packed bf16 conversion.
// ---------------------------------------------------------------------------

#define C512 512
#define NHEAD 8
#define HDIM 64
#define NLAYER 6

typedef __bf16 v16bf __attribute__((ext_vector_type(16)));
typedef float  v8f   __attribute__((ext_vector_type(8)));

__device__ __forceinline__ unsigned short f32_to_bf16(float f) {
    unsigned int u = __float_as_uint(f);
    unsigned int r = u + 0x7FFFu + ((u >> 16) & 1u);   // round-to-nearest-even
    return (unsigned short)(r >> 16);
}

// pack two f32 -> two bf16 (round-to-nearest, ties-away) in one v_perm_b32
__device__ __forceinline__ unsigned int pk_bf16x2(float a, float b) {
    unsigned int ua = __float_as_uint(a) + 0x8000u;
    unsigned int ub = __float_as_uint(b) + 0x8000u;
    // result = { ub[31:16], ua[31:16] } : bytes {7,6,3,2} of {ub,ua}
    return __builtin_amdgcn_perm(ub, ua, 0x07060302u);
}

union FragU {
    v16bf v;
    uint4 q[2];
};

// ---------------------------------------------------------------------------
// Weight pre-pass: W[layer][k][n] f32  ->  Wt[layer][n][k] bf16
// ---------------------------------------------------------------------------
__global__ __launch_bounds__(256)
void convert_transpose_w(const float* __restrict__ W, unsigned short* __restrict__ Wt)
{
    __shared__ float tile[32][33];
    const float* Wl = W + (size_t)blockIdx.z * C512 * C512;
    unsigned short* Wtl = Wt + (size_t)blockIdx.z * C512 * C512;
    int n0 = blockIdx.x * 32, k0 = blockIdx.y * 32;
    int tx = threadIdx.x & 31, ty = threadIdx.x >> 5;   // 32 x 8
    #pragma unroll
    for (int i = 0; i < 4; ++i) {
        int k = k0 + ty + i * 8;
        tile[ty + i * 8][tx] = Wl[(size_t)k * C512 + n0 + tx];
    }
    __syncthreads();
    #pragma unroll
    for (int i = 0; i < 4; ++i) {
        int n = n0 + ty + i * 8;
        Wtl[(size_t)n * C512 + k0 + tx] = f32_to_bf16(tile[tx][ty + i * 8]);
    }
}

// ---------------------------------------------------------------------------
// GEMM: Y[M,512] = X[M,512] @ W[512,512] + bias, optional leaky_relu(0.2).
// Wt is bf16 transposed [n][k]. Block tile 128(M) x 128(N), BK=32, 8 waves
// (4x2), each wave 32x64 (2x4 WMMA tiles). Double-buffered LDS.
// GUARD=false: M known to be a multiple of BM (no bounds checks anywhere).
// ---------------------------------------------------------------------------
template <bool GUARD>
__global__ __launch_bounds__(256)
void gemm512_wmma(const float* __restrict__ X, const unsigned short* __restrict__ Wt,
                  const float* __restrict__ bias, float* __restrict__ Y,
                  int M, int act)
{
    constexpr int BM = 128, BN = 128, BK = 32, LDA = 40; // padded bf16 row stride
    __shared__ __align__(16) unsigned short As[2][BM * LDA];
    __shared__ __align__(16) unsigned short Bs[2][BN * LDA]; // [n][k]

    const int tid  = threadIdx.x;
    const int wave = tid >> 5;
    const int lane = tid & 31;
    const int half = lane >> 4;
    const int l16  = lane & 15;

    const int m0 = blockIdx.x * BM;
    const int n0 = blockIdx.y * BN;
    const int wm = (wave >> 1) * 32;   // 4 wave rows
    const int wn = (wave & 1) * 64;    // 2 wave cols

    // A staging: thread -> (row, 16-wide k chunk)
    const int ar = tid >> 1;
    const int ac = (tid & 1) * 16;
    const int gr = m0 + ar;
    const bool ainb = GUARD ? (gr < M) : true;
    const float* aptr = X + (size_t)gr * C512 + ac;
    // B staging: thread -> (n row, 16-wide k chunk), already bf16
    const int bn = tid >> 1;
    const int bc = (tid & 1) * 16;
    const unsigned short* bptr = Wt + (size_t)(n0 + bn) * C512 + bc;

    auto stage = [&](int k0, int buf) {
        float4 f0, f1, f2, f3;
        if (ainb) {
            const float* p = aptr + k0;
            f0 = *(const float4*)(p);
            f1 = *(const float4*)(p + 4);
            f2 = *(const float4*)(p + 8);
            f3 = *(const float4*)(p + 12);
        } else {
            f0 = f1 = f2 = f3 = make_float4(0.f, 0.f, 0.f, 0.f);
        }
        uint4 lo = make_uint4(pk_bf16x2(f0.x, f0.y), pk_bf16x2(f0.z, f0.w),
                              pk_bf16x2(f1.x, f1.y), pk_bf16x2(f1.z, f1.w));
        uint4 hi = make_uint4(pk_bf16x2(f2.x, f2.y), pk_bf16x2(f2.z, f2.w),
                              pk_bf16x2(f3.x, f3.y), pk_bf16x2(f3.z, f3.w));
        *(uint4*)&As[buf][ar * LDA + ac]     = lo;
        *(uint4*)&As[buf][ar * LDA + ac + 8] = hi;
        const unsigned short* bp = bptr + k0;
        *(uint4*)&Bs[buf][bn * LDA + bc]     = *(const uint4*)(bp);
        *(uint4*)&Bs[buf][bn * LDA + bc + 8] = *(const uint4*)(bp + 8);
        if (k0 + 2 * BK < C512) {   // prefetch next-next tile into caches
            __builtin_prefetch(aptr + k0 + 2 * BK, 0, 0);
            __builtin_prefetch(bptr + k0 + 2 * BK, 0, 0);
        }
    };

    v8f acc[2][4] = {};

    stage(0, 0);
    __syncthreads();

    for (int kb = 0; kb < C512 / BK; ++kb) {
        const int cur = kb & 1;
        if (kb + 1 < C512 / BK) stage((kb + 1) * BK, cur ^ 1);

        v16bf afrag[2], bfrag[4];
        #pragma unroll
        for (int ti = 0; ti < 2; ++ti) {
            int row = wm + ti * 16 + l16;
            const unsigned short* p = &As[cur][row * LDA + half * 8];
            FragU fu;
            fu.q[0] = *(const uint4*)(p);
            fu.q[1] = *(const uint4*)(p + 16);
            afrag[ti] = fu.v;
        }
        #pragma unroll
        for (int tj = 0; tj < 4; ++tj) {
            int nrow = wn + tj * 16 + l16;
            const unsigned short* p = &Bs[cur][nrow * LDA + half * 8];
            FragU fu;
            fu.q[0] = *(const uint4*)(p);
            fu.q[1] = *(const uint4*)(p + 16);
            bfrag[tj] = fu.v;
        }
        #pragma unroll
        for (int ti = 0; ti < 2; ++ti)
            #pragma unroll
            for (int tj = 0; tj < 4; ++tj)
                acc[ti][tj] = __builtin_amdgcn_wmma_f32_16x16x32_bf16(
                    false, afrag[ti], false, bfrag[tj],
                    (short)0, acc[ti][tj], false, false);
        __syncthreads();
    }

    #pragma unroll
    for (int ti = 0; ti < 2; ++ti) {
        #pragma unroll
        for (int tj = 0; tj < 4; ++tj) {
            int col = n0 + wn + tj * 16 + l16;
            float bv = bias[col];
            #pragma unroll
            for (int r = 0; r < 8; ++r) {
                int row = m0 + wm + ti * 16 + r + 8 * half;
                if (!GUARD || row < M) {
                    float o = acc[ti][tj][r] + bv;
                    if (act) o = (o > 0.0f) ? o : o * 0.2f;
                    Y[(size_t)row * C512 + col] = o;
                }
            }
        }
    }
}

// ---------------------------------------------------------------------------
// LayerNorm over rows of width 512. One wave32 per row, 8 rows per block.
// ---------------------------------------------------------------------------
__global__ __launch_bounds__(256)
void ln512_kernel(const float* __restrict__ X, const float* __restrict__ g,
                  const float* __restrict__ b, float* __restrict__ Y, int M)
{
    int wave = threadIdx.x >> 5, lane = threadIdx.x & 31;
    int row = blockIdx.x * 8 + wave;
    if (row >= M) return;
    const float* xr = X + (size_t)row * C512;
    float vals[16];
    float s = 0.0f;
    #pragma unroll
    for (int i = 0; i < 16; ++i) { vals[i] = xr[lane + i * 32]; s += vals[i]; }
    #pragma unroll
    for (int m = 16; m; m >>= 1) s += __shfl_xor(s, m, 32);
    float mean = s * (1.0f / 512.0f);
    float vs = 0.0f;
    #pragma unroll
    for (int i = 0; i < 16; ++i) { float d = vals[i] - mean; vs += d * d; }
    #pragma unroll
    for (int m = 16; m; m >>= 1) vs += __shfl_xor(vs, m, 32);
    float rstd = rsqrtf(vs * (1.0f / 512.0f) + 1e-6f);
    float* yr = Y + (size_t)row * C512;
    #pragma unroll
    for (int i = 0; i < 16; ++i) {
        int c = lane + i * 32;
        yr[c] = (vals[i] - mean) * rstd * g[c] + b[c];
    }
}

// relay init: R[b,c] = mean over l of X[b,l,c]
__global__ void mean_rows_kernel(const float* __restrict__ X, float* __restrict__ R,
                                 int B, int L)
{
    int idx = blockIdx.x * blockDim.x + threadIdx.x;
    if (idx >= B * C512) return;
    int b = idx >> 9, c = idx & 511;
    const float* p = X + ((size_t)b * L) * C512 + c;
    float s = 0.0f;
    for (int l = 0; l < L; ++l) s += p[(size_t)l * C512];
    R[idx] = s / (float)L;
}

// ---------------------------------------------------------------------------
// Ring attention: per (b,l,h), softmax over 5 slots
// {k[l-1], k[l], k[l+1], ek[l], relay_k}; edge keys are zero vectors (score 0).
// One wave32 per (b,l,h); 2 dims per lane (float2).
// ---------------------------------------------------------------------------
__global__ __launch_bounds__(256)
void ring_attn_kernel(const float* __restrict__ Q, const float* __restrict__ Kb,
                      const float* __restrict__ Vb, const float* __restrict__ EK,
                      const float* __restrict__ EV, const float* __restrict__ RK,
                      const float* __restrict__ RV, float* __restrict__ ATT,
                      int B, int L)
{
    int wave = threadIdx.x >> 5, lane = threadIdx.x & 31;
    long gid = (long)blockIdx.x * 8 + wave;          // ((b*L)+l)*NH + h
    long total = (long)B * L * NHEAD;
    if (gid >= total) return;
    int h = (int)(gid & (NHEAD - 1));
    long bl = gid >> 3;                               // b*L + l
    int l = (int)(bl % L);
    int b = (int)(bl / L);

    size_t rowoff = (size_t)bl * C512 + h * HDIM + lane * 2;
    size_t relayoff = (size_t)b * C512 + h * HDIM + lane * 2;

    float2 qv = *(const float2*)(Q + rowoff);

    float2 kv[5], vv[5];
    const float2 z2 = make_float2(0.0f, 0.0f);
    kv[0] = (l > 0)     ? *(const float2*)(Kb + rowoff - C512) : z2;
    kv[1] =               *(const float2*)(Kb + rowoff);
    kv[2] = (l < L - 1) ? *(const float2*)(Kb + rowoff + C512) : z2;
    kv[3] =               *(const float2*)(EK + rowoff);
    kv[4] =               *(const float2*)(RK + relayoff);
    vv[0] = (l > 0)     ? *(const float2*)(Vb + rowoff - C512) : z2;
    vv[1] =               *(const float2*)(Vb + rowoff);
    vv[2] = (l < L - 1) ? *(const float2*)(Vb + rowoff + C512) : z2;
    vv[3] =               *(const float2*)(EV + rowoff);
    vv[4] =               *(const float2*)(RV + relayoff);

    float sc[5];
    #pragma unroll
    for (int w = 0; w < 5; ++w) {
        float p = qv.x * kv[w].x + qv.y * kv[w].y;
        #pragma unroll
        for (int m = 16; m; m >>= 1) p += __shfl_xor(p, m, 32);
        sc[w] = p * 0.125f;   // 1/sqrt(64)
    }
    float mx = sc[0];
    #pragma unroll
    for (int w = 1; w < 5; ++w) mx = fmaxf(mx, sc[w]);
    float sum = 0.0f;
    #pragma unroll
    for (int w = 0; w < 5; ++w) { sc[w] = __expf(sc[w] - mx); sum += sc[w]; }
    float inv = 1.0f / sum;
    float2 out = z2;
    #pragma unroll
    for (int w = 0; w < 5; ++w) {
        out.x += sc[w] * vv[w].x;
        out.y += sc[w] * vv[w].y;
    }
    out.x *= inv; out.y *= inv;
    *(float2*)(ATT + rowoff) = out;
}

// ---------------------------------------------------------------------------
// Star attention: per (b,h), softmax over s=0..L (relay slot + L nodes).
// One workgroup (256 threads) per (b,h).
// ---------------------------------------------------------------------------
__global__ __launch_bounds__(256)
void star_attn_kernel(const float* __restrict__ QR, const float* __restrict__ KYR,
                      const float* __restrict__ VYR, const float* __restrict__ KYN,
                      const float* __restrict__ VYN, float* __restrict__ AR,
                      int B, int L)
{
    __shared__ float sc[2049];
    __shared__ float red[4 * 64];
    __shared__ float rbuf[256];

    int b = blockIdx.x >> 3;
    int h = blockIdx.x & 7;
    int t = threadIdx.x;
    int S = L + 1;

    const float* qr = QR + (size_t)b * C512 + h * HDIM;

    float lmax = -1e30f;
    for (int s = t; s < S; s += 256) {
        const float* kp = (s == 0) ? (KYR + (size_t)b * C512 + h * HDIM)
                                   : (KYN + ((size_t)b * L + (s - 1)) * C512 + h * HDIM);
        float d = 0.0f;
        #pragma unroll
        for (int j = 0; j < HDIM; ++j) d += qr[j] * kp[j];
        d *= 0.125f;
        sc[s] = d;
        lmax = fmaxf(lmax, d);
    }
    rbuf[t] = lmax;
    __syncthreads();
    for (int off = 128; off; off >>= 1) {
        if (t < off) rbuf[t] = fmaxf(rbuf[t], rbuf[t + off]);
        __syncthreads();
    }
    float m = rbuf[0];
    __syncthreads();

    float lsum = 0.0f;
    for (int s = t; s < S; s += 256) {
        float e = __expf(sc[s] - m);
        sc[s] = e;
        lsum += e;
    }
    rbuf[t] = lsum;
    __syncthreads();
    for (int off = 128; off; off >>= 1) {
        if (t < off) rbuf[t] += rbuf[t + off];
        __syncthreads();
    }
    float sum = rbuf[0];
    __syncthreads();

    int d = t & 63, grp = t >> 6;
    float acc = 0.0f;
    for (int s = grp; s < S; s += 4) {
        const float* vp = (s == 0) ? (VYR + (size_t)b * C512 + h * HDIM)
                                   : (VYN + ((size_t)b * L + (s - 1)) * C512 + h * HDIM);
        acc += sc[s] * vp[d];
    }
    red[grp * 64 + d] = acc;
    __syncthreads();
    if (t < 64) {
        float r = (red[t] + red[64 + t] + red[128 + t] + red[192 + t]) / sum;
        AR[(size_t)b * C512 + h * HDIM + t] = r;
    }
}

// ---------------------------------------------------------------------------
// Host orchestration
// ---------------------------------------------------------------------------
extern "C" void kernel_launch(void* const* d_in, const int* in_sizes, int n_in,
                              void* d_out, int out_size, void* d_ws, size_t ws_size,
                              hipStream_t stream)
{
    (void)in_sizes; (void)n_in; (void)out_size; (void)ws_size;

    const int B = 16, L = 2048;
    const int M = B * L;                        // 32768 rows
    const size_t MC = (size_t)M * C512;
    const size_t BC = (size_t)B * C512;
    const size_t WTS = (size_t)NLAYER * C512 * C512;   // bf16 elems per weight tensor

    const float* data = (const float*)d_in[0];
    const float* rq_w = (const float*)d_in[1];  const float* rq_b = (const float*)d_in[2];
    const float* rk_w = (const float*)d_in[3];  const float* rk_b = (const float*)d_in[4];
    const float* rv_w = (const float*)d_in[5];  const float* rv_b = (const float*)d_in[6];
    const float* ro_w = (const float*)d_in[7];  const float* ro_b = (const float*)d_in[8];
    const float* sq_w = (const float*)d_in[9];  const float* sq_b = (const float*)d_in[10];
    const float* sk_w = (const float*)d_in[11]; const float* sk_b = (const float*)d_in[12];
    const float* sv_w = (const float*)d_in[13]; const float* sv_b = (const float*)d_in[14];
    const float* so_w = (const float*)d_in[15]; const float* so_b = (const float*)d_in[16];
    const float* ln_g = (const float*)d_in[17]; const float* ln_b = (const float*)d_in[18];

    float* nodes     = (float*)d_out;           // [M, C] lives in d_out
    float* relay_out = (float*)d_out + MC;      // [B, C]

    float* ws = (float*)d_ws;
    float* XN = ws;            // layernorm output, reused as ATT
    float* Qb = XN + MC;       // q, reused as star ky_nodes
    float* Kb = Qb + MC;       // k, reused as star vy_nodes
    float* Vb = Kb + MC;       // v
    float* EK = Vb + MC;       // ek
    float* EV = EK + MC;       // ev
    float* RELAY = EV + MC;    // [B,C]
    float* RK  = RELAY + BC;
    float* RV  = RK + BC;
    float* QR  = RV + BC;
    float* KYR = QR + BC;
    float* VYR = KYR + BC;
    float* AR  = VYR + BC;
    unsigned short* WT = (unsigned short*)(AR + BC);   // bf16 transposed weights
    unsigned short* wt_rq = WT;
    unsigned short* wt_rk = wt_rq + WTS;
    unsigned short* wt_rv = wt_rk + WTS;
    unsigned short* wt_ro = wt_rv + WTS;
    unsigned short* wt_sq = wt_ro + WTS;
    unsigned short* wt_sk = wt_sq + WTS;
    unsigned short* wt_sv = wt_sk + WTS;
    unsigned short* wt_so = wt_sv + WTS;

    // --- pre-pass: convert+transpose all weights to bf16 [n][k] ---
    {
        dim3 g(C512 / 32, C512 / 32, NLAYER);
        convert_transpose_w<<<g, 256, 0, stream>>>(rq_w, wt_rq);
        convert_transpose_w<<<g, 256, 0, stream>>>(rk_w, wt_rk);
        convert_transpose_w<<<g, 256, 0, stream>>>(rv_w, wt_rv);
        convert_transpose_w<<<g, 256, 0, stream>>>(ro_w, wt_ro);
        convert_transpose_w<<<g, 256, 0, stream>>>(sq_w, wt_sq);
        convert_transpose_w<<<g, 256, 0, stream>>>(sk_w, wt_sk);
        convert_transpose_w<<<g, 256, 0, stream>>>(sv_w, wt_sv);
        convert_transpose_w<<<g, 256, 0, stream>>>(so_w, wt_so);
    }

    auto gemm = [&](const float* X, const unsigned short* Wt, const float* bias,
                    float* Y, int M_, int act) {
        dim3 grid((M_ + 127) / 128, C512 / 128);
        if ((M_ & 127) == 0)
            gemm512_wmma<false><<<grid, 256, 0, stream>>>(X, Wt, bias, Y, M_, act);
        else
            gemm512_wmma<true><<<grid, 256, 0, stream>>>(X, Wt, bias, Y, M_, act);
    };

    // nodes = embs = data ; relay = mean over L
    hipMemcpyAsync(nodes, data, MC * sizeof(float), hipMemcpyDeviceToDevice, stream);
    mean_rows_kernel<<<(B * C512 + 255) / 256, 256, 0, stream>>>(data, RELAY, B, L);

    for (int i = 0; i < NLAYER; ++i) {
        const size_t wo = (size_t)i * C512 * C512;   // per-layer weight offset (elems)
        const size_t bo = (size_t)i * C512;

        // ---- ring attention ----
        ln512_kernel<<<M / 8, 256, 0, stream>>>(nodes, ln_g + bo, ln_b + bo, XN, M);
        gemm(XN,   wt_rq + wo, rq_b + bo, Qb, M, 0);
        gemm(XN,   wt_rk + wo, rk_b + bo, Kb, M, 0);
        gemm(XN,   wt_rv + wo, rv_b + bo, Vb, M, 0);
        gemm(data, wt_rk + wo, rk_b + bo, EK, M, 0);
        gemm(data, wt_rv + wo, rv_b + bo, EV, M, 0);
        gemm(RELAY, wt_rk + wo, rk_b + bo, RK, B, 0);
        gemm(RELAY, wt_rv + wo, rv_b + bo, RV, B, 0);

        ring_attn_kernel<<<(M * NHEAD) / 8, 256, 0, stream>>>(
            Qb, Kb, Vb, EK, EV, RK, RV, XN /*ATT*/, B, L);

        gemm(XN, wt_ro + wo, ro_b + bo, nodes, M, 1);   // leaky relu

        // ---- star attention ----
        gemm(RELAY, wt_sq + wo, sq_b + bo, QR,  B, 0);
        gemm(RELAY, wt_sk + wo, sk_b + bo, KYR, B, 0);
        gemm(RELAY, wt_sv + wo, sv_b + bo, VYR, B, 0);
        gemm(nodes, wt_sk + wo, sk_b + bo, Qb /*KYN*/, M, 0);
        gemm(nodes, wt_sv + wo, sv_b + bo, Kb /*VYN*/, M, 0);

        star_attn_kernel<<<B * NHEAD, 256, 0, stream>>>(
            QR, KYR, VYR, Qb, Kb, AR, B, L);

        gemm(AR, wt_so + wo, so_b + bo, RELAY, B, 1);   // leaky relu
    }

    hipMemcpyAsync(relay_out, RELAY, BC * sizeof(float),
                   hipMemcpyDeviceToDevice, stream);
}
